// EntropyConvD_36472862278329
// MI455X (gfx1250) — matrix operand order repaired and checked by reading, not compiled, not run on hardware
//
#include <hip/hip_runtime.h>
#include <hip/hip_bf16.h>

// ---------------------------------------------------------------------------
// Masked causal 5x5 conv (PixelCNN-style), NCHW f32 -> bf16 implicit GEMM on
// CDNA5 WMMA (v_wmma_f32_16x16x32_bf16), f32 accumulate, bias + leaky ReLU.
// Software-pipelined: double-buffered A (global) / B (LDS) fragments so waits
// are partial and loads overlap the 4-WMMA compute groups.
// ---------------------------------------------------------------------------

typedef __bf16  bf16x8  __attribute__((ext_vector_type(8)));
typedef __bf16  bf16x16 __attribute__((ext_vector_type(16)));
typedef float   f32x8   __attribute__((ext_vector_type(8)));

#define NB     8
#define CCH    128
#define HH     256
#define WW     512
#define NTAPS  13
#define NSTEP  (NTAPS * 4)   // 52 (tap, kb) steps
#define WTILE  64            // output pixels per block (along W)
#define COLS   (WTILE + 4)   // 68 input cols staged
#define ROWS   3             // input rows staged (kh = 0..2)
#define CSTR   136           // padded cin stride: 272B/col -> 4-bank lane step on reads
#define LDS_ELEMS (ROWS * COLS * CSTR)

static __device__ __forceinline__ unsigned short f2bf_rne(float f) {
    unsigned int u = __float_as_uint(f);
    if ((u & 0x7F800000u) == 0x7F800000u && (u & 0x007FFFFFu))
        return (unsigned short)((u >> 16) | 0x0040u);   // quiet NaN
    u += 0x7FFFu + ((u >> 16) & 1u);
    return (unsigned short)(u >> 16);
}

// ---------------------------------------------------------------------------
// Pre-pass: fold mask into weights, convert to bf16, pack into per-lane WMMA
// A-fragment layout (16-bit A 16x32, ISA 7.12.2). id = (cb*52 + s)*32 + lane,
// s = tap*4 + kb; 16 bf16 (32B) per id. Total 425,984 B in d_ws.
// ---------------------------------------------------------------------------
__global__ __launch_bounds__(256) void pack_weights_kernel(
    const float* __restrict__ wgt, unsigned short* __restrict__ apack) {
    int id = blockIdx.x * 256 + threadIdx.x;          // 13312 tuples
    if (id >= 8 * NSTEP * 32) return;
    int lane = id & 31;
    int kb   = (id >> 5) & 3;
    int tap  = (id >> 7) % NTAPS;
    int cb   = id / (128 * NTAPS);

    int kh = (tap < 5) ? 0 : (tap < 10) ? 1 : 2;
    int kw = (tap < 5) ? tap : (tap < 10) ? (tap - 5) : (tap - 10);
    int cout = cb * 16 + (lane & 15);

    unsigned short vals[16];
#pragma unroll
    for (int i = 0; i < 16; ++i) {
        int v = i >> 1, h = i & 1;
        int kp = (v < 4) ? (2 * v + h) : (16 + 2 * (v - 4) + h);
        if (lane >= 16) kp += 8;
        int cin = kb * 32 + kp;
        float f = wgt[((cout * CCH + cin) * 5 + kh) * 5 + kw];
        if (tap == 12 && (cin >> 4) > (cout >> 4)) f = 0.0f;  // HIDDEN center mask
        vals[i] = f2bf_rne(f);
    }
    unsigned short* dst = apack + (size_t)id * 16;
#pragma unroll
    for (int i = 0; i < 16; ++i) dst[i] = vals[i];
}

// ---------------------------------------------------------------------------
// Main kernel: grid (W/64, H, N), block 256 (8 waves). Wave cb owns cout slice
// [cb*16, cb*16+16) x 64 pixels = 4 C tiles; 52 steps x 4 WMMA = 208 WMMAs.
// ---------------------------------------------------------------------------
__global__ __launch_bounds__(256) void entropy_conv_kernel(
    const float* __restrict__ x,
    const unsigned short* __restrict__ apack,
    const float* __restrict__ bias,
    const float* __restrict__ slope,
    float* __restrict__ out) {
    extern __shared__ unsigned short lds[];

    const int tid = threadIdx.x;
    const int w0  = blockIdx.x * WTILE;
    const int h   = blockIdx.y;
    const int n   = blockIdx.z;

    // ---- Stage x window (3 rows x 68 cols x 128 cin) into LDS as bf16 ------
    // quads: (c pair) x (cin pair): one float2 global load per plane, one
    // packed ds_store_b32 per column. ww is always even -> pair-uniform edges.
    const int QTOT = ROWS * (CCH / 2) * (COLS / 2);   // 3*64*34 = 6528
    for (int q = tid; q < QTOT; q += 256) {
        int cp   = q % (COLS / 2);
        int t    = q / (COLS / 2);
        int cinp = t % (CCH / 2);
        int r    = t / (CCH / 2);
        int c    = cp * 2;
        int cin  = cinp * 2;
        int hh   = h + r - 2;
        int ww   = w0 + c - 2;
        int hc   = hh < 0 ? 0 : hh;
        int wc   = ww < 0 ? 0 : (ww > (WW - 2) ? (WW - 2) : ww);
        const float* p0 = x + (((size_t)n * CCH + cin) * HH + hc) * WW + wc;
        float2 a0 = *(const float2*)p0;                       // plane cin
        float2 a1 = *(const float2*)(p0 + (size_t)HH * WW);   // plane cin+1
        bool valid = (hh >= 0) && (ww >= 0) && (ww < WW);
        float f00 = valid ? a0.x : 0.0f, f01 = valid ? a0.y : 0.0f;
        float f10 = valid ? a1.x : 0.0f, f11 = valid ? a1.y : 0.0f;
        unsigned int d0 = (unsigned)f2bf_rne(f00) | ((unsigned)f2bf_rne(f10) << 16);
        unsigned int d1 = (unsigned)f2bf_rne(f01) | ((unsigned)f2bf_rne(f11) << 16);
        *(unsigned int*)(lds + ((r * COLS + c) * CSTR + cin))     = d0;
        *(unsigned int*)(lds + ((r * COLS + c + 1) * CSTR + cin)) = d1;
    }
    __syncthreads();

    const int lane = tid & 31;
    const int cb   = tid >> 5;      // cout block 0..7
    const int ln   = lane & 15;
    const int khi  = lane >> 4;     // which half-wave

    f32x8 acc[4] = {};
    const bf16x16* __restrict__ Afrag = (const bf16x16*)apack;
    const int aBase = cb * (NSTEP * 32) + lane;   // A index = aBase + 32*s

    auto ldA = [&](int s) -> bf16x16 { return Afrag[(size_t)(aBase + 32 * s)]; };

    auto ldB = [&](int s, bf16x8* lo, bf16x8* hi) {
        int tap = s >> 2, kb = s & 3;
        int kh = (tap < 5) ? 0 : (tap < 10) ? 1 : 2;
        int kw = (tap < 5) ? tap : (tap < 10) ? (tap - 5) : (tap - 10);
        int rowBase = kh * COLS + ln + kw;
        int kbase   = kb * 32 + khi * 8;
#pragma unroll
        for (int j = 0; j < 4; ++j) {
            const unsigned short* p = lds + ((rowBase + j * 16) * CSTR + kbase);
            lo[j] = *(const bf16x8*)p;
            hi[j] = *(const bf16x8*)(p + 16);
        }
    };

    auto mma = [&](const bf16x16& a, const bf16x8* lo, const bf16x8* hi) {
#pragma unroll
        for (int j = 0; j < 4; ++j) {
            bf16x16 b = __builtin_shufflevector(lo[j], hi[j],
                    0, 1, 2, 3, 4, 5, 6, 7, 8, 9, 10, 11, 12, 13, 14, 15);
            acc[j] = __builtin_amdgcn_wmma_f32_16x16x32_bf16(
                    false, a, false, b, (short)0, acc[j], false, false);
        }
    };

    // ---- Software-pipelined step loop (ping/pong register double buffer) ---
    bf16x16 aP = ldA(0), aQ;
    bf16x8  loP[4], hiP[4], loQ[4], hiQ[4];
    ldB(0, loP, hiP);
    for (int s = 0; s < NSTEP; s += 2) {
        aQ = ldA(s + 1);
        ldB(s + 1, loQ, hiQ);
        mma(aP, loP, hiP);                       // overlaps with Q loads
        int s2 = (s + 2 < NSTEP) ? s + 2 : s + 1;  // harmless dummy prefetch at end
        aP = ldA(s2);
        ldB(s2, loP, hiP);
        mma(aQ, loQ, hiQ);                       // overlaps with P loads
    }

    // ---- Epilogue: bias + leaky ReLU; C layout: VGPR r -> M = r + 8*khi ----
#pragma unroll
    for (int r = 0; r < 8; ++r) {
        const int co = cb * 16 + r + 8 * khi;
        const float bi = bias[co];
        const float sl = slope[co];
        float* op = out + (((size_t)n * CCH + co) * HH + h) * WW + w0 + ln;
#pragma unroll
        for (int j = 0; j < 4; ++j) {
            float y = acc[j][r] + bi;
            y = (y > 0.0f) ? y : sl * y;
            op[j * 16] = y;
        }
    }
}

extern "C" void kernel_launch(void* const* d_in, const int* in_sizes, int n_in,
                              void* d_out, int out_size, void* d_ws, size_t ws_size,
                              hipStream_t stream) {
    (void)in_sizes; (void)n_in; (void)out_size; (void)ws_size;
    const float* x      = (const float*)d_in[0];
    const float* weight = (const float*)d_in[1];
    const float* bias   = (const float*)d_in[2];
    const float* slope  = (const float*)d_in[3];
    float*       out    = (float*)d_out;
    unsigned short* apack = (unsigned short*)d_ws;  // 425,984 B packed bf16 A frags

    pack_weights_kernel<<<52, 256, 0, stream>>>(weight, apack);

    dim3 grid(WW / WTILE, HH, NB);
    size_t smem = (size_t)LDS_ELEMS * sizeof(unsigned short);  // 55,488 B
    entropy_conv_kernel<<<grid, 256, smem, stream>>>(x, apack, bias, slope, out);
}
